// GPT_66709432041492
// MI455X (gfx1250) — matrix operand order repaired and checked by reading, not compiled, not run on hardware
//
#include <hip/hip_runtime.h>
#include <math.h>

// ---------------------------------------------------------------------------
// GPT block forward for MI455X (gfx1250, wave32, WMMA).
// All matmuls: v_wmma_f32_16x16x32_bf16. Weights converted fp32->bf16 and
// transposed once per launch so both A and B WMMA fragments are contiguous
// 16B/32B per-lane loads using the same contiguous k = 16h + s order in A and B.
// ---------------------------------------------------------------------------

typedef __bf16 bf16_t;
typedef __attribute__((ext_vector_type(8)))  __bf16 v8bf;
typedef __attribute__((ext_vector_type(16))) __bf16 v16bf;
typedef __attribute__((ext_vector_type(8)))  float  v8f;

#define B_ 4
#define T_ 1024
#define C_ 1024
#define H_ 16
#define D_ 64
#define L_ 12
#define M_ (B_ * T_) /* 4096 rows in the token dimension */

// round-to-nearest-even fp32 -> bf16 without relying on __bf16 arithmetic
__device__ __forceinline__ bf16_t f2bf(float f) {
  unsigned u = __builtin_bit_cast(unsigned, f);
  unsigned r = u + 0x7fffu + ((u >> 16) & 1u);
  return __builtin_bit_cast(bf16_t, (unsigned short)(r >> 16));
}

__device__ __forceinline__ v8f wmma_bf16(v16bf a, v16bf b, v8f c) {
  // (neg_a, A, neg_b, B, c_mod, C, reuse_a, reuse_b)
  return __builtin_amdgcn_wmma_f32_16x16x32_bf16(false, a, false, b,
                                                 (short)0, c, false, false);
}

// A-matrix 16x32 bf16 fragment: lane half h holds K range
// [16h, 16h+16) of its row (matches B) -> two contiguous 16B loads.
__device__ __forceinline__ v16bf load_afrag(const bf16_t* rowp, int off) {
  v8bf lo = *(const v8bf*)(rowp + off);
  v8bf hi = *(const v8bf*)(rowp + off + 8);
  return __builtin_shufflevector(lo, hi, 0, 1, 2, 3, 4, 5, 6, 7,
                                 8, 9, 10, 11, 12, 13, 14, 15);
}

// ---------------------------------------------------------------------------
// Weight convert + transpose: W[l][K][N] fp32 -> WT[l][N][K] bf16.
// Coalesced read (n fastest), strided write; runs once per launch.
// ---------------------------------------------------------------------------
__global__ void wconv_kernel(const float* __restrict__ W, bf16_t* __restrict__ WT,
                             int Kd, int Nd, long long total) {
  long long i = (long long)blockIdx.x * 256 + threadIdx.x;
  if (i >= total) return;
  int per = Kd * Nd;
  int l = (int)(i / per);
  int r = (int)(i % per);
  int k = r / Nd, n = r % Nd;
  WT[(size_t)l * per + (size_t)n * Kd + k] = f2bf(W[i]);
}

__global__ void copy_kernel(const float* __restrict__ src, float* __restrict__ dst, int n) {
  int i = blockIdx.x * 256 + threadIdx.x;
  if (i < n) dst[i] = src[i];
}

// ---------------------------------------------------------------------------
// LayerNorm: one 256-thread block per row of C=1024. Optional bf16 and/or
// fp32 outputs (bf16 feeds the next WMMA GEMM; fp32 for the final output).
// ---------------------------------------------------------------------------
__global__ __launch_bounds__(256) void ln_kernel(const float* __restrict__ x,
                                                 const float* __restrict__ g,
                                                 const float* __restrict__ bt,
                                                 bf16_t* __restrict__ obf,
                                                 float* __restrict__ of) {
  __shared__ float sh1[256], sh2[256];
  const int row = blockIdx.x;
  const float* xr = x + (size_t)row * C_;
  float s = 0.f, s2 = 0.f;
  for (int i = threadIdx.x; i < C_; i += 256) {
    float v = xr[i];
    s += v;
    s2 += v * v;
  }
  sh1[threadIdx.x] = s;
  sh2[threadIdx.x] = s2;
  __syncthreads();
  for (int off = 128; off > 0; off >>= 1) {
    if (threadIdx.x < off) {
      sh1[threadIdx.x] += sh1[threadIdx.x + off];
      sh2[threadIdx.x] += sh2[threadIdx.x + off];
    }
    __syncthreads();
  }
  const float mean = sh1[0] * (1.f / C_);
  const float var = sh2[0] * (1.f / C_) - mean * mean;
  const float rs = rsqrtf(var + 1e-5f);
  for (int i = threadIdx.x; i < C_; i += 256) {
    float v = (xr[i] - mean) * rs * g[i] + bt[i];
    if (obf) obf[(size_t)row * C_ + i] = f2bf(v);
    if (of) of[(size_t)row * C_ + i] = v;
  }
}

// ---------------------------------------------------------------------------
// WMMA GEMM: out[M][N] = A[M][K] @ W[K][N] + bias (+ residual).
// A: bf16 row-major. WT: bf16 [N][K] (pre-transposed weights).
// One wave -> 16x64 strip (4 N-tiles sharing the A fragment). 8 waves/block.
// ---------------------------------------------------------------------------
__global__ __launch_bounds__(256) void gemm_kernel(
    const bf16_t* __restrict__ A, const bf16_t* __restrict__ WT,
    const float* __restrict__ bias, const float* __restrict__ residual,
    float* __restrict__ outF, bf16_t* __restrict__ outBF, int M, int N, int K) {
  const int lane = threadIdx.x & 31;
  const int gwave = blockIdx.x * 8 + (threadIdx.x >> 5);
  const int ntiles = N >> 6;
  const int mt = gwave / ntiles;
  const int nt = gwave % ntiles;
  if (mt >= (M >> 4)) return;  // wave-uniform exit, EXEC stays full for WMMA
  const int m0 = mt << 4, n0 = nt << 6;
  const int half = lane >> 4, l16 = lane & 15;

  const bf16_t* Arow = A + (size_t)(m0 + l16) * K;
  // B-matrix 32x16 bf16 fragment: lane half h holds K rows [16h,16h+16) of
  // column l16 -> with WT[N][K] this is one contiguous 32B load per lane.
  const bf16_t* B0 = WT + (size_t)(n0 + 0 + l16) * K + half * 16;
  const bf16_t* B1 = WT + (size_t)(n0 + 16 + l16) * K + half * 16;
  const bf16_t* B2 = WT + (size_t)(n0 + 32 + l16) * K + half * 16;
  const bf16_t* B3 = WT + (size_t)(n0 + 48 + l16) * K + half * 16;

  v8f c0 = {}, c1 = {}, c2 = {}, c3 = {};
  for (int k0 = 0; k0 < K; k0 += 32) {
    __builtin_prefetch(B0 + k0 + 256, 0, 1);  // global_prefetch_b8 for the weight stream
    v16bf a = load_afrag(Arow, k0 + half * 16);
    v16bf b0 = *(const v16bf*)(B0 + k0);
    v16bf b1 = *(const v16bf*)(B1 + k0);
    v16bf b2 = *(const v16bf*)(B2 + k0);
    v16bf b3 = *(const v16bf*)(B3 + k0);
    c0 = wmma_bf16(a, b0, c0);
    c1 = wmma_bf16(a, b1, c1);
    c2 = wmma_bf16(a, b2, c2);
    c3 = wmma_bf16(a, b3, c3);
  }

  // C/D layout: lane = column n0+l16; VGPR v = row v (+8 for upper lane half)
#pragma unroll
  for (int v = 0; v < 8; ++v) {
    const int row = m0 + v + (half << 3);
    const size_t base = (size_t)row * N + n0 + l16;
    float vals[4] = {c0[v], c1[v], c2[v], c3[v]};
#pragma unroll
    for (int t = 0; t < 4; ++t) {
      const size_t idx = base + t * 16;
      float val = vals[t] + bias[n0 + t * 16 + l16];
      if (residual) val += residual[idx];
      if (outF) outF[idx] = val;
      if (outBF) outBF[idx] = f2bf(val);
    }
  }
}

// ---------------------------------------------------------------------------
// Split fused QKV (fp32, bias already added) into bf16 head-major tensors:
// Q[B,H,T,D], K[B,H,T,D], and V transposed as VT[B,H,D,T] so the P@V
// B-fragments load contiguously.
// ---------------------------------------------------------------------------
__global__ void splitqkv_kernel(const float* __restrict__ qkv, bf16_t* __restrict__ Q,
                                bf16_t* __restrict__ K, bf16_t* __restrict__ VT) {
  int i = blockIdx.x * 256 + threadIdx.x;
  if (i >= B_ * T_ * C_) return;
  int c = i & (C_ - 1);
  int t = (i >> 10) & (T_ - 1);
  int b = i >> 20;
  int h = c >> 6, d = c & 63;
  size_t src = (size_t)(b * T_ + t) * (3 * C_) + c;
  size_t bh = (size_t)(b * H_ + h);
  Q[(bh * T_ + t) * D_ + d] = f2bf(qkv[src]);
  K[(bh * T_ + t) * D_ + d] = f2bf(qkv[src + C_]);
  VT[(bh * D_ + d) * T_ + t] = f2bf(qkv[src + 2 * C_]);
}

// ---------------------------------------------------------------------------
// Flash-style causal attention. One wave (blockDim=32 => barrier == S_NOP)
// per (b,h,q-tile of 16). K-tile = 32 keys: S via 2x2 WMMAs (D=64), online
// softmax with shfl reductions over 16-lane halves, P restaged via LDS into
// the A-fragment layout, O += P@V via 4 WMMAs against VT.
// ---------------------------------------------------------------------------
__global__ __launch_bounds__(32) void attn_kernel(const bf16_t* __restrict__ Qh,
                                                  const bf16_t* __restrict__ Kh,
                                                  const bf16_t* __restrict__ VTh,
                                                  bf16_t* __restrict__ Y) {
  __shared__ __align__(32) bf16_t Psh[16 * 32];
  const int lane = threadIdx.x & 31;
  const int half = lane >> 4, l16 = lane & 15;
  const int qt = blockIdx.x & 63;   // T/16
  const int bh = blockIdx.x >> 6;   // b*H + h
  const int q0 = qt << 4;

  const bf16_t* Qb = Qh + (size_t)bh * T_ * D_;
  const bf16_t* Kb = Kh + (size_t)bh * T_ * D_;
  const bf16_t* Vb = VTh + (size_t)bh * D_ * T_;

  // Q fragments for the full D=64 contraction (two K=32 chunks), loaded once
  const bf16_t* Qrow = Qb + (size_t)(q0 + l16) * D_;
  const v16bf qa0 = load_afrag(Qrow, half * 16);
  const v16bf qa1 = load_afrag(Qrow, half * 16 + 32);

  v8f o0 = {}, o1 = {}, o2 = {}, o3 = {};
  float m[8], l[8];
#pragma unroll
  for (int v = 0; v < 8; ++v) {
    m[v] = -1e30f;
    l[v] = 0.f;
  }
  const float scale = 0.125f;  // 1/sqrt(64)

  for (int k0 = 0; k0 <= q0 + 15; k0 += 32) {
    const bool have2 = (k0 + 16 <= q0 + 15);  // uniform

    // S = Q @ K^T for keys [k0, k0+16) and [k0+16, k0+32)
    const bf16_t* Kr0 = Kb + (size_t)(k0 + l16) * D_ + half * 16;
    v8f s0 = {};
    s0 = wmma_bf16(qa0, *(const v16bf*)(Kr0), s0);
    s0 = wmma_bf16(qa1, *(const v16bf*)(Kr0 + 32), s0);
    v8f s1 = {};
    if (have2) {
      const bf16_t* Kr1 = Kb + (size_t)(k0 + 16 + l16) * D_ + half * 16;
      s1 = wmma_bf16(qa0, *(const v16bf*)(Kr1), s1);
      s1 = wmma_bf16(qa1, *(const v16bf*)(Kr1 + 32), s1);
    }

    // online softmax over the 32-key tile; rows live in (VGPR v, lane half)
#pragma unroll
    for (int v = 0; v < 8; ++v) {
      const int rglob = q0 + v + (half << 3);
      float sv0 = (k0 + l16 <= rglob) ? s0[v] * scale : -1e30f;
      float sv1 = (have2 && (k0 + 16 + l16 <= rglob)) ? s1[v] * scale : -1e30f;
      float rm = fmaxf(sv0, sv1);
      rm = fmaxf(rm, __shfl_xor(rm, 1));
      rm = fmaxf(rm, __shfl_xor(rm, 2));
      rm = fmaxf(rm, __shfl_xor(rm, 4));
      rm = fmaxf(rm, __shfl_xor(rm, 8));
      const float mn = fmaxf(m[v], rm);
      const float alpha = __expf(m[v] - mn);
      const float p0 = __expf(sv0 - mn);
      const float p1 = __expf(sv1 - mn);
      float rs = p0 + p1;
      rs += __shfl_xor(rs, 1);
      rs += __shfl_xor(rs, 2);
      rs += __shfl_xor(rs, 4);
      rs += __shfl_xor(rs, 8);
      l[v] = l[v] * alpha + rs;
      m[v] = mn;
      o0[v] *= alpha;
      o1[v] *= alpha;
      o2[v] *= alpha;
      o3[v] *= alpha;
      const int r32 = (v + (half << 3)) * 32;
      Psh[r32 + l16] = f2bf(p0);        // all 512 P entries rewritten
      Psh[r32 + 16 + l16] = f2bf(p1);   // each iteration (no pre-zero)
    }
    __syncthreads();  // single-wave WG: S_NOP + compiler fence

    // P (16x32, C-layout in LDS row-major) reloaded as an A fragment
    const v16bf pa = load_afrag(&Psh[l16 * 32], half * 16);
    // V B-fragments from VT[D][T]: contiguous 32B per lane
    const bf16_t* Vr = Vb + (size_t)l16 * T_ + k0 + half * 16;
    o0 = wmma_bf16(pa, *(const v16bf*)(Vr + 0 * 16 * T_), o0);
    o1 = wmma_bf16(pa, *(const v16bf*)(Vr + 1 * 16 * T_), o1);
    o2 = wmma_bf16(pa, *(const v16bf*)(Vr + 2 * 16 * T_), o2);
    o3 = wmma_bf16(pa, *(const v16bf*)(Vr + 3 * 16 * T_), o3);
    __syncthreads();
  }

  // write O / l back as bf16 rows of Y[B,T,C]
  const int b = bh >> 4, h = bh & 15;
#pragma unroll
  for (int v = 0; v < 8; ++v) {
    const int row = q0 + v + (half << 3);
    const float inv = 1.f / l[v];
    const size_t base = ((size_t)(b * T_ + row)) * C_ + h * D_ + l16;
    Y[base + 0] = f2bf(o0[v] * inv);
    Y[base + 16] = f2bf(o1[v] * inv);
    Y[base + 32] = f2bf(o2[v] * inv);
    Y[base + 48] = f2bf(o3[v] * inv);
  }
}

__global__ void gelu_kernel(const float* __restrict__ in, bf16_t* __restrict__ out, int n) {
  int i = blockIdx.x * 256 + threadIdx.x;
  if (i >= n) return;
  float v = in[i];
  float t = tanhf(0.7978845608028654f * (v + 0.044715f * v * v * v));
  out[i] = f2bf(0.5f * v * (1.f + t));
}

// ---------------------------------------------------------------------------
// Host orchestration
// ---------------------------------------------------------------------------
extern "C" void kernel_launch(void* const* d_in, const int* in_sizes, int n_in,
                              void* d_out, int out_size, void* d_ws, size_t ws_size,
                              hipStream_t stream) {
  (void)in_sizes; (void)n_in; (void)out_size; (void)ws_size;
  const float* x = (const float*)d_in[0];
  const float* ln1_g = (const float*)d_in[1];
  const float* ln1_b = (const float*)d_in[2];
  const float* attn_w = (const float*)d_in[3];
  const float* attn_b = (const float*)d_in[4];
  const float* proj_w = (const float*)d_in[5];
  const float* proj_b = (const float*)d_in[6];
  const float* ln2_g = (const float*)d_in[7];
  const float* ln2_b = (const float*)d_in[8];
  const float* fc_w = (const float*)d_in[9];
  const float* fc_b = (const float*)d_in[10];
  const float* fc2_w = (const float*)d_in[11];
  const float* fc2_b = (const float*)d_in[12];
  const float* lnf_g = (const float*)d_in[13];
  const float* lnf_b = (const float*)d_in[14];

  // workspace bump allocator (256B aligned)
  char* p = (char*)d_ws;
  auto alloc = [&](size_t bytes) -> void* {
    void* r = (void*)p;
    p += (bytes + 255) & ~(size_t)255;
    return r;
  };
  float* xres = (float*)alloc((size_t)M_ * C_ * 4);            // residual stream, fp32
  bf16_t* hbuf = (bf16_t*)alloc((size_t)M_ * C_ * 2);          // LN output, bf16
  float* qkv = (float*)alloc((size_t)M_ * 3 * C_ * 4);         // fused QKV, fp32
  bf16_t* Qh = (bf16_t*)alloc((size_t)M_ * C_ * 2);
  bf16_t* Kh = (bf16_t*)alloc((size_t)M_ * C_ * 2);
  bf16_t* VTh = (bf16_t*)alloc((size_t)M_ * C_ * 2);
  bf16_t* ybuf = (bf16_t*)alloc((size_t)M_ * C_ * 2);          // attention out
  float* fcout = (float*)alloc((size_t)M_ * 4 * C_ * 4);       // FC pre-GELU
  bf16_t* h2 = (bf16_t*)alloc((size_t)M_ * 4 * C_ * 2);        // GELU out
  bf16_t* attnWT = (bf16_t*)alloc((size_t)L_ * 3 * C_ * C_ * 2);
  bf16_t* projWT = (bf16_t*)alloc((size_t)L_ * C_ * C_ * 2);
  bf16_t* fcWT = (bf16_t*)alloc((size_t)L_ * 4 * C_ * C_ * 2);
  bf16_t* fc2WT = (bf16_t*)alloc((size_t)L_ * C_ * 4 * C_ * 2);

  auto launch_wconv = [&](const float* W, bf16_t* WT, int Kd, int Nd) {
    long long total = (long long)L_ * Kd * Nd;
    unsigned blocks = (unsigned)((total + 255) / 256);
    wconv_kernel<<<blocks, 256, 0, stream>>>(W, WT, Kd, Nd, total);
  };
  auto launch_gemm = [&](const bf16_t* A, const bf16_t* WT, const float* bias,
                         const float* res, float* oF, bf16_t* oBF, int M, int N, int K) {
    int waves = (M / 16) * (N / 64);
    gemm_kernel<<<(waves + 7) / 8, 256, 0, stream>>>(A, WT, bias, res, oF, oBF, M, N, K);
  };
  auto launch_ln = [&](const float* in, const float* g, const float* b,
                       bf16_t* obf, float* of) {
    ln_kernel<<<M_, 256, 0, stream>>>(in, g, b, obf, of);
  };

  // one-time (per launch) weight conversion + transpose
  launch_wconv(attn_w, attnWT, C_, 3 * C_);
  launch_wconv(proj_w, projWT, C_, C_);
  launch_wconv(fc_w, fcWT, C_, 4 * C_);
  launch_wconv(fc2_w, fc2WT, 4 * C_, C_);
  copy_kernel<<<(M_ * C_ + 255) / 256, 256, 0, stream>>>(x, xres, M_ * C_);

  for (int layer = 0; layer < L_; ++layer) {
    // ---- attention half ----
    launch_ln(xres, ln1_g + layer * C_, ln1_b + layer * C_, hbuf, nullptr);
    launch_gemm(hbuf, attnWT + (size_t)layer * 3 * C_ * C_, attn_b + layer * 3 * C_,
                nullptr, qkv, nullptr, M_, 3 * C_, C_);
    splitqkv_kernel<<<(M_ * C_ + 255) / 256, 256, 0, stream>>>(qkv, Qh, Kh, VTh);
    attn_kernel<<<B_ * H_ * (T_ / 16), 32, 0, stream>>>(Qh, Kh, VTh, ybuf);
    launch_gemm(ybuf, projWT + (size_t)layer * C_ * C_, proj_b + layer * C_,
                xres, xres, nullptr, M_, C_, C_);
    // ---- MLP half ----
    launch_ln(xres, ln2_g + layer * C_, ln2_b + layer * C_, hbuf, nullptr);
    launch_gemm(hbuf, fcWT + (size_t)layer * 4 * C_ * C_, fc_b + layer * 4 * C_,
                nullptr, fcout, nullptr, M_, 4 * C_, C_);
    gelu_kernel<<<(M_ * 4 * C_ + 255) / 256, 256, 0, stream>>>(fcout, h2, M_ * 4 * C_);
    launch_gemm(h2, fc2WT + (size_t)layer * C_ * 4 * C_, fc2_b + layer * C_,
                xres, xres, nullptr, M_, C_, 4 * C_);
  }

  // final LayerNorm -> fp32 output
  launch_ln(xres, lnf_g, lnf_b, nullptr, (float*)d_out);
}
